// MemoryAggregator_695784702254
// MI455X (gfx1250) — matrix operand order repaired and checked by reading, not compiled, not run on hardware
//
#include <hip/hip_runtime.h>
#include <hip/hip_bf16.h>

// ---------------------------------------------------------------------------
// Graph attention aggregation for MI455X (gfx1250, wave32, WMMA + async LDS)
//   Q = X@Wq, K = X@Wk, V = X@Wv           (bf16 WMMA, fp32 accumulate)
//   s_e = <Q[src],K[dst]>/8 ; per-src softmax ; out[src] += alpha_e * V[dst]
// Normalization deferred: accumulate Sum(e*V) and Sum(e), divide once.
// ---------------------------------------------------------------------------

typedef __attribute__((ext_vector_type(16))) __bf16 bf16x16;
typedef __attribute__((ext_vector_type(8)))  __bf16 bf16x8;
typedef __attribute__((ext_vector_type(8)))  float  f32x8;

union FragAB { bf16x16 v; bf16x8 h[2]; };

#define TILE_ROWS 32          // X rows per block (2 waves x 16)
#define LDS_STRIDE 264        // bf16 elems/row: 528 B == 16 B mod 256 -> conflict-free b128 LDS reads
#define HDIM 64
#define DDIM 256

// order-preserving float <-> uint map for atomic segment-max
__device__ __forceinline__ unsigned f2key(float f) {
    unsigned b = __float_as_uint(f);
    return (b & 0x80000000u) ? ~b : (b | 0x80000000u);
}
__device__ __forceinline__ float key2f(unsigned k) {
    unsigned b = (k & 0x80000000u) ? (k & 0x7FFFFFFFu) : ~k;
    return __uint_as_float(b);
}

// async global->LDS 16-byte copy (ASYNCcnt-tracked, GV addressing mode)
__device__ __forceinline__ void async_copy_b128(const void* gptr, void* lptr) {
    unsigned lds_addr = (unsigned)(size_t)lptr;              // LDS address is the low 32 bits of the pointer
    unsigned long long gaddr = (unsigned long long)(size_t)gptr;
    asm volatile("global_load_async_to_lds_b128 %0, %1, off"
                 :: "v"(lds_addr), "v"(gaddr) : "memory");
}
__device__ __forceinline__ void wait_async0() {
    asm volatile("s_wait_asynccnt 0" ::: "memory");
}

// ---------------------------------------------------------------------------
// Kernel 1: init out / denom / max-keys
// ---------------------------------------------------------------------------
__global__ void init_kernel(float* __restrict__ out, float* __restrict__ denom,
                            unsigned* __restrict__ mkey, int n_nodes) {
    long long t = (long long)blockIdx.x * blockDim.x + threadIdx.x;
    long long total = (long long)n_nodes * HDIM;
    if (t < total) out[t] = 0.0f;
    if (t < n_nodes) { denom[t] = 0.0f; mkey[t] = 0u; }
}

// ---------------------------------------------------------------------------
// Kernel 2: fused Q/K/V GEMM via v_wmma_f32_16x16x32_bf16
//   block = 64 threads (2 waves), each wave owns 16 rows of the 32-row tile.
//   Global data staged through LDS with global_load_async_to_lds_b128.
// ---------------------------------------------------------------------------
__global__ void qkv_wmma_kernel(const float* __restrict__ X,
                                const float* __restrict__ Wq,
                                const float* __restrict__ Wk,
                                const float* __restrict__ Wv,
                                float* __restrict__ Q,
                                float* __restrict__ K,
                                float* __restrict__ V,
                                int n_nodes) {
    __shared__ __align__(16) float  stagef[2048];                 // 8 KB async landing buffer
    __shared__ __align__(16) __bf16 Xs[TILE_ROWS * LDS_STRIDE];   // X tile bf16, row major
    __shared__ __align__(16) __bf16 Ws[HDIM      * LDS_STRIDE];   // W^T bf16 (col major), reused

    const int tid    = threadIdx.x;
    const int row0   = blockIdx.x * TILE_ROWS;
    const int lane   = tid & 31;
    const int wave   = tid >> 5;         // 0..1
    const int lane16 = lane & 15;
    const int half   = lane >> 4;        // 0: lanes 0-15, 1: lanes 16-31

    // ---- stage X tile: async fp32 -> stage, convert -> bf16 Xs (4 chunks of 8 rows) ----
    for (int c = 0; c < 4; ++c) {
        const int rbase = c * 8;
        for (int f = tid; f < 512; f += blockDim.x) {       // 512 float4 per 8-row chunk
            int r  = f >> 6;                                // 0..7
            int gr = row0 + rbase + r;
            if (gr < n_nodes)
                async_copy_b128(X + (size_t)gr * DDIM + (f & 63) * 4, &stagef[f * 4]);
        }
        wait_async0();
        __syncthreads();
        for (int i = tid; i < 8 * DDIM; i += blockDim.x) {
            int r = i >> 8, k = i & 255;
            int gr = row0 + rbase + r;
            Xs[(rbase + r) * LDS_STRIDE + k] =
                (gr < n_nodes) ? (__bf16)stagef[i] : (__bf16)0.0f;
        }
        __syncthreads();
    }

    const float* Wsrc[3] = {Wq, Wk, Wv};
    float*       Osel[3] = {Q, K, V};

    for (int wsel = 0; wsel < 3; ++wsel) {
        // ---- stage W^T: async fp32 -> stage, convert transposed -> bf16 Ws ----
        const float* W = Wsrc[wsel];
        for (int c = 0; c < 8; ++c) {                       // 8 chunks of 32 W-rows (k)
            const int kbase = c * 32;
            for (int f = tid; f < 512; f += blockDim.x)     // 32 rows x 16 float4
                async_copy_b128(W + (size_t)(kbase + (f >> 4)) * HDIM + (f & 15) * 4,
                                &stagef[f * 4]);
            wait_async0();
            __syncthreads();    // also fences previous weight's WMMA reads of Ws
            for (int i = tid; i < 32 * HDIM; i += blockDim.x) {
                int r = i >> 6, n = i & 63;
                Ws[n * LDS_STRIDE + kbase + r] = (__bf16)stagef[i];
            }
            __syncthreads();
        }

        // ---- WMMA: 16 rows x 64 cols per wave, K = 256 (full EXEC, no divergence) ----
        f32x8 acc[4] = {{}, {}, {}, {}};
        const int rowBase = (wave * 16 + lane16) * LDS_STRIDE;
#pragma unroll
        for (int ks = 0; ks < DDIM / 32; ++ks) {
            const int k0 = ks * 32;
            // A frag (ISA 16-bit A layout: lanes 0-15 K[0..7],[16..23]; 16-31 shifted by 8)
            FragAB a;
            a.h[0] = *(const bf16x8*)&Xs[rowBase + k0 + half * 8];
            a.h[1] = *(const bf16x8*)&Xs[rowBase + k0 + 16 + half * 8];
#pragma unroll
            for (int n = 0; n < 4; ++n) {
                // B frag (same K order as A: lanes 0-15 K[0..7],[16..23]; 16-31 shifted by 8)
                FragAB b;
                const int cb = (n * 16 + lane16) * LDS_STRIDE + k0 + half * 8;
                b.h[0] = *(const bf16x8*)&Ws[cb];
                b.h[1] = *(const bf16x8*)&Ws[cb + 16];
                acc[n] = __builtin_amdgcn_wmma_f32_16x16x32_bf16(
                    false, a.v, false, b.v, (short)0, acc[n], false, false);
            }
        }

        // ---- store C (f32 16x16 layout: VGPR v, lane -> M = v + half*8, N = lane16) ----
        float* O = Osel[wsel];
#pragma unroll
        for (int n = 0; n < 4; ++n) {
#pragma unroll
            for (int v = 0; v < 8; ++v) {
                int m    = v + half * 8;
                int node = row0 + wave * 16 + m;
                if (node < n_nodes)
                    O[(size_t)node * HDIM + n * 16 + lane16] = acc[n][v];
            }
        }
    }
}

// ---------------------------------------------------------------------------
// Kernel 3: edge scores + atomic segment max (8 lanes per edge)
// ---------------------------------------------------------------------------
__global__ void edge_scores_kernel(const float* __restrict__ Q,
                                   const float* __restrict__ K,
                                   const long long* __restrict__ ei,
                                   float* __restrict__ scores,
                                   unsigned* __restrict__ mkey,
                                   int n_edges) {
    long long t = (long long)blockIdx.x * blockDim.x + threadIdx.x;
    int e   = (int)(t >> 3);
    int sub = (int)(t & 7);
    if (e >= n_edges) return;
    long long s = 0, d = 0;
    if (sub == 0) { s = ei[e]; d = ei[(long long)n_edges + e]; }
    s = __shfl(s, 0, 8);                   // broadcast within the 8-lane edge group
    d = __shfl(d, 0, 8);
    const float4* q = (const float4*)(Q + (size_t)s * HDIM);
    const float4* k = (const float4*)(K + (size_t)d * HDIM);
    float acc = 0.0f;
    float4 a = q[sub],     b = k[sub];
    acc += a.x * b.x + a.y * b.y + a.z * b.z + a.w * b.w;
    a = q[sub + 8];        b = k[sub + 8];
    acc += a.x * b.x + a.y * b.y + a.z * b.z + a.w * b.w;
    acc += __shfl_xor(acc, 1, 32);
    acc += __shfl_xor(acc, 2, 32);
    acc += __shfl_xor(acc, 4, 32);
    if (sub == 0) {
        float sc = acc * 0.125f;           // / sqrt(64)
        scores[e] = sc;
        atomicMax(&mkey[s], f2key(sc));
    }
}

// ---------------------------------------------------------------------------
// Kernel 4: per-edge e = exp(s - m) (in-place over scores) + denom atomic
// ---------------------------------------------------------------------------
__global__ void edge_exp_kernel(float* __restrict__ scores,
                                const unsigned* __restrict__ mkey,
                                const long long* __restrict__ ei,
                                float* __restrict__ denom,
                                int n_edges) {
    long long e = (long long)blockIdx.x * blockDim.x + threadIdx.x;
    if (e >= n_edges) return;
    long long s = ei[e];
    float ev = __expf(scores[e] - key2f(mkey[s]));
    scores[e] = ev;
    atomicAdd(&denom[s], ev);
}

// ---------------------------------------------------------------------------
// Kernel 5: out[src] += e * V[dst]   (one wave per edge, broadcast indices,
//           wave-uniform base addresses -> coalesced loads + L2 atomics)
// ---------------------------------------------------------------------------
__global__ void edge_accum_kernel(const float* __restrict__ V,
                                  const float* __restrict__ scores,
                                  const long long* __restrict__ ei,
                                  float* __restrict__ out,
                                  int n_edges) {
    int wave = threadIdx.x >> 5;
    int lane = threadIdx.x & 31;
    int e = blockIdx.x * 8 + wave;                  // 8 waves per block
    if (e >= n_edges) return;
    long long s = 0, d = 0; float ev = 0.0f;
    if (lane == 0) { s = ei[e]; d = ei[(long long)n_edges + e]; ev = scores[e]; }
    s  = __shfl(s, 0);
    d  = __shfl(d, 0);
    ev = __shfl(ev, 0);
    const float* vp = V + (size_t)d * HDIM;
    float*       op = out + (size_t)s * HDIM;
    atomicAdd(&op[lane],      ev * vp[lane]);
    atomicAdd(&op[lane + 32], ev * vp[lane + 32]);
}

// ---------------------------------------------------------------------------
// Kernel 6: out[i] /= max(denom[i], 1e-38)   (empty rows stay exactly 0)
// ---------------------------------------------------------------------------
__global__ void normalize_kernel(float* __restrict__ out,
                                 const float* __restrict__ denom,
                                 int n_nodes) {
    long long t = (long long)blockIdx.x * blockDim.x + threadIdx.x;
    if (t >= (long long)n_nodes * HDIM) return;
    out[t] /= fmaxf(denom[t >> 6], 1e-38f);
}

// ---------------------------------------------------------------------------
extern "C" void kernel_launch(void* const* d_in, const int* in_sizes, int n_in,
                              void* d_out, int out_size, void* d_ws, size_t ws_size,
                              hipStream_t stream) {
    const float*     X  = (const float*)d_in[0];
    const float*     Wq = (const float*)d_in[1];
    const float*     Wk = (const float*)d_in[2];
    const float*     Wv = (const float*)d_in[3];
    const long long* ei = (const long long*)d_in[4];

    const int n_nodes = in_sizes[0] / DDIM;   // 100000
    const int n_edges = in_sizes[4] / 2;      // 1600000

    // workspace layout: Q | K | V | scores | mkey | denom  (~84 MB total)
    float*    Q      = (float*)d_ws;
    float*    K      = Q + (size_t)n_nodes * HDIM;
    float*    V      = K + (size_t)n_nodes * HDIM;
    float*    scores = V + (size_t)n_nodes * HDIM;
    unsigned* mkey   = (unsigned*)(scores + n_edges);
    float*    denom  = (float*)(mkey + n_nodes);
    float*    out    = (float*)d_out;

    {   // init
        long long total = (long long)n_nodes * HDIM;
        init_kernel<<<(int)((total + 255) / 256), 256, 0, stream>>>(out, denom, mkey, n_nodes);
    }
    {   // Q/K/V GEMM (WMMA bf16, async LDS staging)
        int blocks = (n_nodes + TILE_ROWS - 1) / TILE_ROWS;
        qkv_wmma_kernel<<<blocks, 64, 0, stream>>>(X, Wq, Wk, Wv, Q, K, V, n_nodes);
    }
    {   // scores + segment max
        long long total = (long long)n_edges * 8;
        edge_scores_kernel<<<(int)((total + 255) / 256), 256, 0, stream>>>(Q, K, ei, scores, mkey, n_edges);
    }
    {   // per-edge exp + denom
        edge_exp_kernel<<<(n_edges + 255) / 256, 256, 0, stream>>>(scores, mkey, ei, denom, n_edges);
    }
    {   // weighted V scatter (one wave per edge)
        edge_accum_kernel<<<(n_edges + 7) / 8, 256, 0, stream>>>(V, scores, ei, out, n_edges);
    }
    {   // normalize
        long long total = (long long)n_nodes * HDIM;
        normalize_kernel<<<(int)((total + 255) / 256), 256, 0, stream>>>(out, denom, n_nodes);
    }
}
